// EDLSTM_1D_tf_34462817583779
// MI455X (gfx1250) — matrix operand order repaired and checked by reading, not compiled, not run on hardware
//
#include <hip/hip_runtime.h>

// Encoder-decoder LSTM, fully fused persistent kernel for gfx1250 (wave32, WMMA).
// B=4096 split into 256 workgroups x 16 batch rows; each WG runs the whole
// 168-step 2-layer encoder + 24-step decoder with h/c state in LDS and
// per-wave weight fragments in registers. Gate GEMMs use v_wmma_f32_16x16x32_f16.
// All per-sequence data (label_p, time indices) is staged into LDS up front so
// the sequential recurrence never touches global memory; nonlinearities use
// hardware v_tanh_f32 / v_rcp_f32 instead of IEEE divides.

#define ENCT 168
#define DECT 24
#define HH   64
#define NG   256   // 4*H gate width
#define INK  20    // encoder/decoder input feature width (padded to 32)
#define BT   16    // batch rows per workgroup (one WMMA M tile)
#define HSTR 72    // padded f16 row stride for h buffers (16B-aligned rows)

typedef __attribute__((ext_vector_type(16))) _Float16 v16h;
typedef __attribute__((ext_vector_type(8)))  _Float16 v8h;
typedef __attribute__((ext_vector_type(8)))  float    v8f;

#if __has_builtin(__builtin_amdgcn_tanhf)
__device__ __forceinline__ float mytanh(float x) { return __builtin_amdgcn_tanhf(x); }
#else
__device__ __forceinline__ float mytanh(float x) {
  float e = __expf(2.f * x);                       // v_exp_f32
  return 1.f - 2.f * __builtin_amdgcn_rcpf(e + 1.f);  // v_rcp_f32, no div_scale
}
#endif
__device__ __forceinline__ float sigm(float x) {
  return fmaf(0.5f, mytanh(0.5f * x), 0.5f);
}

// A-matrix (16x32 f16, M x K) fragment from LDS, row-major rows of `stride` halves.
// ISA layout: lanes 0-15 -> M=lane, K in {klo..klo+7, klo+16..klo+23}, klo = (lane>=16)*8.
__device__ __forceinline__ v16h load_A_lds(const _Float16* base, int stride, int lane) {
  int row = lane & 15;
  int klo = (lane >> 4) << 3;
  const _Float16* p = base + row * stride + klo;
  v8h lo = *(const v8h*)(p);        // K = klo .. klo+7
  v8h hi = *(const v8h*)(p + 16);   // K = klo+16 .. klo+23
  v16h a;
#pragma unroll
  for (int i = 0; i < 8; ++i) { a[i] = lo[i]; a[i + 8] = hi[i]; }
  return a;
}

// B-matrix (32x16 f16, K x N) fragment built from a row-major f32 weight W[N][ld]
// (we need W^T). ISA layout: N = lane%16, K in {klo..klo+7, klo+16..klo+23}, klo = (lane>=16)*8 (+kbase).
__device__ __forceinline__ v16h load_B_w(const float* W, int ld, int kvalid,
                                         int nbase, int kbase, int lane) {
  int n = nbase + (lane & 15);
  int klo = kbase + ((lane >> 4) << 3);
  v16h b;
#pragma unroll
  for (int r = 0; r < 8; ++r) {
    int k0 = klo + 2 * r + (r >= 4 ? 8 : 0);
    int k1 = k0 + 1;
    float f0 = (k0 < kvalid) ? W[n * ld + k0] : 0.f;
    float f1 = (k1 < kvalid) ? W[n * ld + k1] : 0.f;
    b[2 * r]     = (_Float16)f0;
    b[2 * r + 1] = (_Float16)f1;
  }
  return b;
}

__device__ __forceinline__ v8f wmma_step(v16h a, v16h b, v8f c) {
  return __builtin_amdgcn_wmma_f32_16x16x32_f16(false, a, false, b, (short)0, c,
                                                false, false);
}

// C/D layout: VGPR r -> M = (lane>=16?8:0)+r, N = lane%16.
__device__ __forceinline__ void store_acc(float* gbuf, v8f acc, int ntile, int lane) {
  int ncol  = ntile * 16 + (lane & 15);
  int mbase = (lane >> 4) << 3;
#pragma unroll
  for (int r = 0; r < 8; ++r) gbuf[(mbase + r) * NG + ncol] = acc[r];
}

// Elementwise LSTM cell update: 256 threads cover 16x64 (h,c) elements.
__device__ __forceinline__ void lstm_elem(int tid, const float* g, float* cb,
                                          _Float16* hb) {
  int m  = tid >> 4;
  int jb = (tid & 15) * 4;
#pragma unroll
  for (int q = 0; q < 4; ++q) {
    int j = jb + q;
    float gi = g[m * NG + j];
    float gf = g[m * NG + 64 + j];
    float gc = g[m * NG + 128 + j];
    float go = g[m * NG + 192 + j];
    float c  = cb[m * HH + j];
    c = sigm(gf) * c + sigm(gi) * mytanh(gc);
    cb[m * HH + j]   = c;
    hb[m * HSTR + j] = (_Float16)(sigm(go) * mytanh(c));
  }
}

__global__ __launch_bounds__(256) void edlstm_fused_kernel(
    const int* __restrict__ input_time, const float* __restrict__ label_p,
    const int* __restrict__ decoder_time,
    const float* __restrict__ W0i, const float* __restrict__ W0h,
    const float* __restrict__ b0i, const float* __restrict__ b0h,
    const float* __restrict__ W1i, const float* __restrict__ W1h,
    const float* __restrict__ b1i, const float* __restrict__ b1h,
    const float* __restrict__ Wdi, const float* __restrict__ Wdh,
    const float* __restrict__ bdi, const float* __restrict__ bdh,
    const float* __restrict__ Wout, const float* __restrict__ bout,
    const float* __restrict__ eh, const float* __restrict__ ew,
    const float* __restrict__ em, const float* __restrict__ ey,
    float* __restrict__ out) {
  __shared__ __align__(16) _Float16 xbuf[BT * 32];
  __shared__ __align__(16) _Float16 hbuf0[BT * HSTR];
  __shared__ __align__(16) _Float16 hbuf1[BT * HSTR];
  __shared__ __align__(16) float cbuf0[BT * HH];
  __shared__ __align__(16) float cbuf1[BT * HH];
  __shared__ __align__(16) float gbuf[BT * NG];
  __shared__ float semb[574];   // hour[0..119] wd[120..143] mo[144..195] woy[196..573]
  __shared__ float sWout[HH];
  __shared__ float sprev[BT];
  __shared__ float sbout;
  // Staged per-WG sequence data: recurrence loop never touches global memory.
  __shared__ float         sLabel[BT * ENCT];        // 10.5 KB
  __shared__ unsigned char sItime[BT * ENCT * 4];    // 10.5 KB (indices < 54)
  __shared__ unsigned char sDtime[BT * DECT * 4];    // 1.5 KB

  const int tid  = threadIdx.x;
  const int lane = tid & 31;
  const int wv   = tid >> 5;            // wave 0..7 -> owns N tiles 2wv, 2wv+1
  const int B0   = blockIdx.x * BT;

  // ---- one-time preload: embeddings, Wout, sequence data; zero h/c state ----
  for (int i = tid; i < 120; i += 256) semb[i]       = eh[i];
  for (int i = tid; i < 24;  i += 256) semb[120 + i] = ew[i];
  for (int i = tid; i < 52;  i += 256) semb[144 + i] = em[i];
  for (int i = tid; i < 378; i += 256) semb[196 + i] = ey[i];
  if (tid < HH) sWout[tid] = Wout[tid];
  if (tid == 64) sbout = bout[0];
  for (int i = tid; i < BT * ENCT; i += 256) {
    int row = i / ENCT, t = i % ENCT;
    sLabel[i] = label_p[(B0 + row) * ENCT + t];
  }
  for (int i = tid; i < BT * ENCT * 4; i += 256) {
    int row = i / (ENCT * 4), rem = i % (ENCT * 4);
    sItime[i] = (unsigned char)input_time[(B0 + row) * ENCT * 4 + rem];
  }
  for (int i = tid; i < BT * DECT * 4; i += 256) {
    int row = i / (DECT * 4), rem = i % (DECT * 4);
    sDtime[i] = (unsigned char)decoder_time[(B0 + row) * DECT * 4 + rem];
  }
  for (int i = tid; i < BT * HSTR; i += 256) {
    hbuf0[i] = (_Float16)0.f;
    hbuf1[i] = (_Float16)0.f;
  }
  for (int i = tid; i < BT * HH; i += 256) { cbuf0[i] = 0.f; cbuf1[i] = 0.f; }

  // ---- encoder weight fragments: register-resident, reused 168x ----
  const int n0 = wv * 2 * 16;
  v16h B0x[2], B0h[2][2], B1x[2][2], B1h[2][2];
  float bias0[2], bias1[2];
#pragma unroll
  for (int ts = 0; ts < 2; ++ts) {
    int nb = n0 + ts * 16;
    B0x[ts]    = load_B_w(W0i, INK, INK, nb, 0, lane);
    B0h[ts][0] = load_B_w(W0h, HH, HH, nb, 0, lane);
    B0h[ts][1] = load_B_w(W0h, HH, HH, nb, 32, lane);
    B1x[ts][0] = load_B_w(W1i, HH, HH, nb, 0, lane);
    B1x[ts][1] = load_B_w(W1i, HH, HH, nb, 32, lane);
    B1h[ts][0] = load_B_w(W1h, HH, HH, nb, 0, lane);
    B1h[ts][1] = load_B_w(W1h, HH, HH, nb, 32, lane);
    int nn    = nb + (lane & 15);
    bias0[ts] = b0i[nn] + b0h[nn];
    bias1[ts] = b1i[nn] + b1h[nn];
  }
  __syncthreads();

  // ---- encoder: 168 steps, two fused LSTM layers, all-LDS inner loop ----
  for (int t = 0; t < ENCT; ++t) {
    // build x(t) = [label_p, time embeddings], zero-padded K 20..31
    for (int e = tid; e < BT * 32; e += 256) {
      int row = e >> 5, col = e & 31;
      float v = 0.f;
      if (col == 0) {
        v = sLabel[row * ENCT + t];
      } else if (col < 20) {
        const unsigned char* tp = sItime + (row * ENCT + t) * 4;
        int c = col - 1;
        if (c < 5)       v = semb[tp[0] * 5 + c];
        else if (c < 8)  v = semb[120 + tp[1] * 3 + (c - 5)];
        else if (c < 12) v = semb[144 + tp[2] * 4 + (c - 8)];
        else             v = semb[196 + tp[3] * 7 + (c - 12)];
      }
      xbuf[row * 32 + col] = (_Float16)v;
    }
    __syncthreads();

    {  // layer 0 gates: x(t)@W0i^T + h0(t-1)@W0h^T + bias
      v16h ax  = load_A_lds(xbuf, 32, lane);
      v16h ah0 = load_A_lds(hbuf0, HSTR, lane);
      v16h ah1 = load_A_lds(hbuf0 + 32, HSTR, lane);
#pragma unroll
      for (int ts = 0; ts < 2; ++ts) {
        v8f acc;
#pragma unroll
        for (int r = 0; r < 8; ++r) acc[r] = bias0[ts];
        acc = wmma_step(ax,  B0x[ts],    acc);
        acc = wmma_step(ah0, B0h[ts][0], acc);
        acc = wmma_step(ah1, B0h[ts][1], acc);
        store_acc(gbuf, acc, wv * 2 + ts, lane);
      }
    }
    __syncthreads();
    lstm_elem(tid, gbuf, cbuf0, hbuf0);   // -> h0(t), c0(t)
    __syncthreads();

    {  // layer 1 gates: h0(t)@W1i^T + h1(t-1)@W1h^T + bias
      v16h ax0 = load_A_lds(hbuf0, HSTR, lane);
      v16h ax1 = load_A_lds(hbuf0 + 32, HSTR, lane);
      v16h ah0 = load_A_lds(hbuf1, HSTR, lane);
      v16h ah1 = load_A_lds(hbuf1 + 32, HSTR, lane);
#pragma unroll
      for (int ts = 0; ts < 2; ++ts) {
        v8f acc;
#pragma unroll
        for (int r = 0; r < 8; ++r) acc[r] = bias1[ts];
        acc = wmma_step(ax0, B1x[ts][0], acc);
        acc = wmma_step(ax1, B1x[ts][1], acc);
        acc = wmma_step(ah0, B1h[ts][0], acc);
        acc = wmma_step(ah1, B1h[ts][1], acc);
        store_acc(gbuf, acc, wv * 2 + ts, lane);
      }
    }
    __syncthreads();
    lstm_elem(tid, gbuf, cbuf1, hbuf1);   // -> h1(t), c1(t)
    __syncthreads();
  }

  // ---- decoder weight fragments ----
  v16h Bdx[2], Bdh[2][2];
  float biasd[2];
#pragma unroll
  for (int ts = 0; ts < 2; ++ts) {
    int nb = n0 + ts * 16;
    Bdx[ts]    = load_B_w(Wdi, INK, INK, nb, 0, lane);
    Bdh[ts][0] = load_B_w(Wdh, HH, HH, nb, 0, lane);
    Bdh[ts][1] = load_B_w(Wdh, HH, HH, nb, 32, lane);
    int nn    = nb + (lane & 15);
    biasd[ts] = bdi[nn] + bdh[nn];
  }
  if (tid < BT) sprev[tid] = sLabel[tid * ENCT + (ENCT - 1)];
  __syncthreads();

  // ---- decoder: 24 autoregressive steps, state continues in hbuf1/cbuf1 ----
  for (int t = 0; t < DECT; ++t) {
    for (int e = tid; e < BT * 32; e += 256) {
      int row = e >> 5, col = e & 31;
      float v = 0.f;
      if (col == 0) {
        v = sprev[row];
      } else if (col < 20) {
        const unsigned char* tp = sDtime + (row * DECT + t) * 4;
        int c = col - 1;
        if (c < 5)       v = semb[tp[0] * 5 + c];
        else if (c < 8)  v = semb[120 + tp[1] * 3 + (c - 5)];
        else if (c < 12) v = semb[144 + tp[2] * 4 + (c - 8)];
        else             v = semb[196 + tp[3] * 7 + (c - 12)];
      }
      xbuf[row * 32 + col] = (_Float16)v;
    }
    __syncthreads();

    {
      v16h ax  = load_A_lds(xbuf, 32, lane);
      v16h ah0 = load_A_lds(hbuf1, HSTR, lane);
      v16h ah1 = load_A_lds(hbuf1 + 32, HSTR, lane);
#pragma unroll
      for (int ts = 0; ts < 2; ++ts) {
        v8f acc;
#pragma unroll
        for (int r = 0; r < 8; ++r) acc[r] = biasd[ts];
        acc = wmma_step(ax,  Bdx[ts],    acc);
        acc = wmma_step(ah0, Bdh[ts][0], acc);
        acc = wmma_step(ah1, Bdh[ts][1], acc);
        store_acc(gbuf, acc, wv * 2 + ts, lane);
      }
    }
    __syncthreads();
    lstm_elem(tid, gbuf, cbuf1, hbuf1);
    __syncthreads();

    if (tid < BT) {  // out = h1 @ Wout^T + bout ; feeds next step
      float s = sbout;
#pragma unroll 8
      for (int k = 0; k < HH; ++k) s += (float)hbuf1[tid * HSTR + k] * sWout[k];
      out[(B0 + tid) * DECT + t] = s;
      sprev[tid] = s;
    }
    __syncthreads();
  }
}

extern "C" void kernel_launch(void* const* d_in, const int* in_sizes, int n_in,
                              void* d_out, int out_size, void* d_ws, size_t ws_size,
                              hipStream_t stream) {
  // setup_inputs order:
  // 0 input_p_q (unused) 1 input_time 2 label_p 3 decoder_time
  // 4..7 enc0 Wih/Whh/bih/bhh  8..11 enc1  12..15 dec  16 Wout 17 bout
  // 18 emb_hour 19 emb_wd 20 emb_mo 21 emb_woy
  const int*   input_time   = (const int*)d_in[1];
  const float* label_p      = (const float*)d_in[2];
  const int*   decoder_time = (const int*)d_in[3];
  const float* W0i = (const float*)d_in[4];
  const float* W0h = (const float*)d_in[5];
  const float* b0i = (const float*)d_in[6];
  const float* b0h = (const float*)d_in[7];
  const float* W1i = (const float*)d_in[8];
  const float* W1h = (const float*)d_in[9];
  const float* b1i = (const float*)d_in[10];
  const float* b1h = (const float*)d_in[11];
  const float* Wdi = (const float*)d_in[12];
  const float* Wdh = (const float*)d_in[13];
  const float* bdi = (const float*)d_in[14];
  const float* bdh = (const float*)d_in[15];
  const float* Wout = (const float*)d_in[16];
  const float* bout = (const float*)d_in[17];
  const float* eh = (const float*)d_in[18];
  const float* ew = (const float*)d_in[19];
  const float* em = (const float*)d_in[20];
  const float* ey = (const float*)d_in[21];
  float* out = (float*)d_out;

  int Bn = in_sizes[2] / ENCT;   // 4096
  int grid = Bn / BT;            // 256 workgroups, 16 batch rows each
  edlstm_fused_kernel<<<grid, 256, 0, stream>>>(
      input_time, label_p, decoder_time, W0i, W0h, b0i, b0h, W1i, W1h, b1i, b1h,
      Wdi, Wdh, bdi, bdh, Wout, bout, eh, ew, em, ey, out);
}
